// WITRAN_2DPSGMU_Encoder_36867999269073
// MI455X (gfx1250) — matrix operand order repaired and checked by reading, not compiled, not run on hardware
//
#include <hip/hip_runtime.h>
#include <hip/hip_bf16.h>
#include <math.h>

// ---------------- problem constants ----------------
#define Bn   64     // batch
#define Od   24     // Original_slice_len (input dim1)
#define Rd   24     // Water2sea_slice_num (input dim2)
#define Id   128    // input size / embed dim
#define Hd   256    // hidden
#define Ld   47     // Rd + Od - 1
#define RB   (Rd*Bn)          // 1536 rows per step
#define MALL ((size_t)Ld*RB)  // 72192

#define OUT_ALL ((size_t)RB*Ld*2*Hd)     // 36,962,304 floats
#define COLSZ   ((size_t)Bn*Od*Hd)       // 393,216 floats

typedef __attribute__((ext_vector_type(16))) _Float16 v16h;
typedef __attribute__((ext_vector_type(8)))  float    v8f;

// ---------------- WMMA helpers (CDNA5 16x16x32 f16 -> f32) ----------------
__device__ inline v8f wmma16(v16h a, v16h b, v8f c) {
  return __builtin_amdgcn_wmma_f32_16x16x32_f16(false, a, false, b, (short)0, c,
                                                false, false);
}

// A tile in LDS: 16x32 row-major (row stride 32). Lane layout used here (contiguous k, same in fragB/fragBt):
// lane: M = lane&15, hi = lane>>4; elem e: j=e>>1,p=e&1; K = 2j + 16*hi + p
__device__ inline v16h fragA(const _Float16* s, int lane) {
  v16h a; int m = lane & 15, hi = lane >> 4;
#pragma unroll
  for (int e = 0; e < 16; ++e) {
    int j = e >> 1, p = e & 1;
    int kk = 2*j + 16*hi + p;
    a[e] = s[m*32 + kk];
  }
  return a;
}
// B tile in LDS stored k-major 32x16 (row stride 16): B[kk][n] = s[kk*16+n]
__device__ inline v16h fragB(const _Float16* s, int lane) {
  v16h b; int n = lane & 15, hi = lane >> 4;
#pragma unroll
  for (int e = 0; e < 16; ++e) {
    int j = e >> 1, p = e & 1;
    int kk = 2*j + 16*hi + p;
    b[e] = s[kk*16 + n];
  }
  return b;
}
// B tile in LDS stored n-major 16x32 (row stride 32): B[kk][n] = s[n*32+kk]
__device__ inline v16h fragBt(const _Float16* s, int lane) {
  v16h b; int n = lane & 15, hi = lane >> 4;
#pragma unroll
  for (int e = 0; e < 16; ++e) {
    int j = e >> 1, p = e & 1;
    int kk = 2*j + 16*hi + p;
    b[e] = s[n*32 + kk];
  }
  return b;
}

__device__ inline float sigmf(float x) { return 1.f / (1.f + __expf(-x)); }

// ---------------- elementwise kernels ----------------
__global__ void k_cvt_f16(const float* __restrict__ src, _Float16* __restrict__ dst, int n) {
  int i = blockIdx.x * 256 + threadIdx.x;
  if (i < n) dst[i] = (_Float16)src[i];
}

// a_steps[l,r,b,i] = (0 <= l-r < O) ? input[b, l-r, r, i] : 0   (store f16)
__global__ void k_build_x(const float* __restrict__ inp, _Float16* __restrict__ xh) {
  size_t idx = (size_t)blockIdx.x * 256 + threadIdx.x;   // L*R*B*I total, exact grid
  int i = idx % Id; size_t t = idx / Id;
  int b = t % Bn;  t /= Bn;
  int r = t % Rd;  int l = (int)(t / Rd);
  int o = l - r;
  float v = (o >= 0 && o < Od) ? inp[((((size_t)b*Od + o)*Rd + r)*Id) + i] : 0.f;
  xh[idx] = (_Float16)v;
}

// ------- GEMM: C = A(MxK) @ Wt(NxK)^T + bias, f16 out.  M,N % 64 == 0, K % 32 == 0.
// 128 threads = 4 waves; block tile 64x64; wave w owns rows [16w,16w+16) x all 64 cols.
__global__ void k_gemm_aw(const _Float16* __restrict__ A, const _Float16* __restrict__ Wt,
                          const float* __restrict__ bias, _Float16* __restrict__ Cout,
                          int M, int N, int K) {
  __shared__ __align__(16) _Float16 sA[64*32];
  __shared__ __align__(16) _Float16 sB[64*32];     // n-major
  int nb = N >> 6;
  int mb = blockIdx.x / nb, nbk = blockIdx.x % nb;
  int tid = threadIdx.x, wave = tid >> 5, lane = tid & 31;
  v8f acc[4] = {};
  for (int k0 = 0; k0 < K; k0 += 32) {
#pragma unroll
    for (int c = tid; c < 256; c += 128) {          // 8-half (16B) chunks
      int row = c >> 2, kc = (c & 3) << 3;
      *(uint4*)&sA[row*32 + kc] = *(const uint4*)&A [(size_t)(mb*64 + row)*K + k0 + kc];
      *(uint4*)&sB[row*32 + kc] = *(const uint4*)&Wt[(size_t)(nbk*64 + row)*K + k0 + kc];
    }
    __syncthreads();
    v16h af = fragA(&sA[wave*16*32], lane);
#pragma unroll
    for (int nn = 0; nn < 4; ++nn)
      acc[nn] = wmma16(af, fragBt(&sB[nn*16*32], lane), acc[nn]);
    __syncthreads();
  }
  int n = lane & 15, hi = lane >> 4;
#pragma unroll
  for (int nn = 0; nn < 4; ++nn)
#pragma unroll
    for (int e = 0; e < 8; ++e) {
      int m  = mb*64 + wave*16 + e + 8*hi;
      int ng = nbk*64 + nn*16 + n;
      Cout[(size_t)m * N + ng] = (_Float16)(acc[nn][e] + bias[ng]);
    }
}

// ---------------- attention: scores = q@k^T/8 + mask -----------------------
// qkv layout: [(l*R + r)*B + b]*384 + {q:0,k:128,v:256} + h*64 + d
__global__ void k_scores(const _Float16* __restrict__ qkv, const float* __restrict__ mask,
                         float* __restrict__ sc) {
  __shared__ _Float16 sA[512];
  __shared__ _Float16 sB[512];
  int sub = blockIdx.x & 3, lbh = blockIdx.x >> 2;
  int mt = sub >> 1, nt = sub & 1;
  int h = lbh & 1; int lb = lbh >> 1; int b = lb % Bn; int l = lb / Bn;
  int lane = threadIdx.x;
  v8f acc = {};
  for (int k0 = 0; k0 < 64; k0 += 32) {
    for (int t = lane; t < 512; t += 32) {
      int m = t >> 5, kk = t & 31;
      int row = mt*16 + m;
      sA[t] = (row < Rd) ? qkv[(((size_t)l*Rd + row)*Bn + b)*384 + h*64 + k0 + kk]
                         : (_Float16)0.f;
      int kb = t >> 4, n = t & 15;
      int col = nt*16 + n;
      sB[t] = (col < Rd) ? qkv[(((size_t)l*Rd + col)*Bn + b)*384 + 128 + h*64 + k0 + kb]
                         : (_Float16)0.f;
    }
    __syncthreads();
    acc = wmma16(fragA(sA, lane), fragB(sB, lane), acc);
    __syncthreads();
  }
  int n = lane & 15, hi = lane >> 4;
#pragma unroll
  for (int e = 0; e < 8; ++e) {
    int row = mt*16 + e + 8*hi, col = nt*16 + n;
    if (row < Rd && col < Rd)
      sc[(size_t)lbh*576 + row*24 + col] = acc[e] * 0.125f + mask[row*24 + col];
  }
}

// softmax over last axis (24), one wave per row; write f16 attn
__global__ void k_softmax(const float* __restrict__ sc, _Float16* __restrict__ attnh) {
  int lbh = blockIdx.x / 24, r = blockIdx.x % 24;
  size_t base = (size_t)lbh*576 + r*24;
  int lane = threadIdx.x;
  float v = (lane < 24) ? sc[base + lane] : -1e30f;
  float m = v;
#pragma unroll
  for (int o = 16; o; o >>= 1) m = fmaxf(m, __shfl_xor(m, o));
  float e = (lane < 24) ? __expf(v - m) : 0.f;
  float s = e;
#pragma unroll
  for (int o = 16; o; o >>= 1) s += __shfl_xor(s, o);
  if (lane < 24) attnh[base + lane] = (_Float16)(e / s);
}

// context: c[r, h*64+d] = sum_s attn[r,s] * v[s,d]   (M=24, N=64, K=24 pad 32)
__global__ void k_ctx(const _Float16* __restrict__ attnh, const _Float16* __restrict__ qkv,
                      _Float16* __restrict__ ch) {
  __shared__ _Float16 sA[512];
  __shared__ _Float16 sB[512];
  int sub = blockIdx.x & 7, lbh = blockIdx.x >> 3;
  int mt = sub >> 2, nt = sub & 3;
  int h = lbh & 1; int lb = lbh >> 1; int b = lb % Bn; int l = lb / Bn;
  int lane = threadIdx.x;
  for (int t = lane; t < 512; t += 32) {
    int m = t >> 5, kk = t & 31;
    int row = mt*16 + m;
    sA[t] = (row < Rd && kk < 24) ? attnh[(size_t)lbh*576 + row*24 + kk] : (_Float16)0.f;
    int kb = t >> 4, n = t & 15;
    sB[t] = (kb < 24) ? qkv[(((size_t)l*Rd + kb)*Bn + b)*384 + 256 + h*64 + nt*16 + n]
                      : (_Float16)0.f;
  }
  __syncthreads();
  v8f acc = {};
  acc = wmma16(fragA(sA, lane), fragB(sB, lane), acc);
  int n = lane & 15, hi = lane >> 4;
#pragma unroll
  for (int e = 0; e < 8; ++e) {
    int row = mt*16 + e + 8*hi;
    if (row < Rd)
      ch[(((size_t)l*Rd + row)*Bn + b)*128 + h*64 + nt*16 + n] = (_Float16)acc[e];
  }
}

// ------- recurrence GEMM: g = [h_row,h_col,x,y] @ W^T (+bias cond), f32 out.
// M=N=1536, K=768. Block tile 64x64 (4 waves), A gathered from 4 concat sources.
__global__ void k_gemm_g(const _Float16* __restrict__ hrowh, const _Float16* __restrict__ hcolh,
                         const _Float16* __restrict__ xh_l, const _Float16* __restrict__ yh_l,
                         const _Float16* __restrict__ Wh, const float* __restrict__ Bias,
                         float* __restrict__ g, int step) {
  __shared__ __align__(16) _Float16 sA[64*32];
  __shared__ __align__(16) _Float16 sB[64*32];     // n-major
  int mb = blockIdx.x / 24, nbk = blockIdx.x % 24;
  int tid = threadIdx.x, wave = tid >> 5, lane = tid & 31;
  v8f acc[4] = {};
  for (int k0 = 0; k0 < 768; k0 += 32) {
#pragma unroll
    for (int c = tid; c < 256; c += 128) {
      int row = c >> 2, kc = (c & 3) << 3;
      int mg = mb*64 + row;
      int k  = k0 + kc;
      const _Float16* src;
      if (k < 256)      src = &hrowh[(size_t)mg*Hd + k];
      else if (k < 512) src = &hcolh[(size_t)mg*Hd + (k - 256)];
      else if (k < 640) src = &xh_l [(size_t)mg*Id + (k - 512)];
      else              src = &yh_l [(size_t)mg*Id + (k - 640)];
      *(uint4*)&sA[row*32 + kc] = *(const uint4*)src;
      *(uint4*)&sB[row*32 + kc] = *(const uint4*)&Wh[(size_t)(nbk*64 + row)*768 + k];
    }
    __syncthreads();
    v16h af = fragA(&sA[wave*16*32], lane);
#pragma unroll
    for (int nn = 0; nn < 4; ++nn)
      acc[nn] = wmma16(af, fragBt(&sB[nn*16*32], lane), acc[nn]);
    __syncthreads();
  }
  int n = lane & 15, hi = lane >> 4;
#pragma unroll
  for (int nn = 0; nn < 4; ++nn)
#pragma unroll
    for (int e = 0; e < 8; ++e) {
      int mg = mb*64 + wave*16 + e + 8*hi;
      int r  = mg >> 6;               // row index (B = 64)
      int ng = nbk*64 + nn*16 + n;
      float v = acc[nn][e];
      if (r <= step && step < Rd) v += Bias[ng];
      g[(size_t)mg*1536 + ng] = v;
    }
}

// gates + state update + output scatter + h_col roll (ping-pong)
__global__ void k_gates(const float* __restrict__ g, float* __restrict__ hrow,
                        const float* __restrict__ hcol_in, float* __restrict__ hcol_out,
                        _Float16* __restrict__ hrowh, _Float16* __restrict__ hcol_outh,
                        float* __restrict__ out, int l) {
  int idx = blockIdx.x * 256 + threadIdx.x;   // RB*H exact
  int j = idx & 255; int m = idx >> 8;
  int r = m >> 6, b = m & 63;
  const float* gm = g + (size_t)m * 1536;
  float ug_r = sigmf(gm[j]);
  float og_r = sigmf(gm[256 + j]);
  float ug_c = sigmf(gm[512 + j]);
  float og_c = sigmf(gm[768 + j]);
  float ig_r = tanhf(gm[1024 + j]);
  float ig_c = tanhf(gm[1280 + j]);
  float hr = tanhf((1.f - ug_r) * hrow[idx]    + ug_r * ig_r) * og_r;
  float hc = tanhf((1.f - ug_c) * hcol_in[idx] + ug_c * ig_c) * og_c;
  hrow[idx] = hr; hrowh[idx] = (_Float16)hr;
  int rr = (r + 1) % Rd;                           // roll by +1 along R
  size_t oidx = ((size_t)rr*Bn + b)*Hd + j;
  hcol_out[oidx] = hc; hcol_outh[oidx] = (_Float16)hc;
  // output_all[(r*B+b), l, 0:256]=hr, [256:512]=hc
  size_t ob = (size_t)m * (Ld*2*Hd) + (size_t)l * (2*Hd);
  out[ob + j] = hr;
  out[ob + Hd + j] = hc;
  if (l >= Od - 1) {
    int t = l - (Od - 1);
    if (r == Rd - 1) out[OUT_ALL + ((size_t)b*Od + t)*Hd + j] = hc;          // hidden_col_all
    if (r == t)      out[OUT_ALL + COLSZ + ((size_t)b*Rd + t)*Hd + j] = hr;  // hidden_row_all
  }
}

// ---------------- host side ----------------
extern "C" void kernel_launch(void* const* d_in, const int* in_sizes, int n_in,
                              void* d_out, int out_size, void* d_ws, size_t ws_size,
                              hipStream_t stream) {
  (void)in_sizes; (void)n_in; (void)out_size; (void)ws_size;
  const float* inp  = (const float*)d_in[0];
  const float* mask = (const float*)d_in[1];
  const float* W    = (const float*)d_in[2];
  const float* Bias = (const float*)d_in[3];
  const float* ipw  = (const float*)d_in[4];
  const float* ipb  = (const float*)d_in[5];
  const float* opw  = (const float*)d_in[6];
  const float* opb  = (const float*)d_in[7];
  float* out = (float*)d_out;

  // workspace carve-up (256B aligned)
  char* p = (char*)d_ws;
  auto carve = [&](size_t bytes) { void* r = (void*)p; p += (bytes + 255) & ~(size_t)255; return r; };
  _Float16* Wh    = (_Float16*)carve((size_t)1536*768 * 2);
  _Float16* ipwh  = (_Float16*)carve((size_t)384*128  * 2);
  _Float16* opwh  = (_Float16*)carve((size_t)128*128  * 2);
  _Float16* xh    = (_Float16*)carve(MALL*Id  * 2);
  _Float16* qkvh  = (_Float16*)carve(MALL*384 * 2);
  float*    sc    = (float*)   carve((size_t)Ld*Bn*2*576 * 4);
  _Float16* attnh = (_Float16*)carve((size_t)Ld*Bn*2*576 * 2);
  _Float16* ch    = (_Float16*)carve(MALL*Id  * 2);
  _Float16* yh    = (_Float16*)carve(MALL*Id  * 2);
  float*    hrow  = (float*)   carve((size_t)RB*Hd * 4);
  _Float16* hrowh = (_Float16*)carve((size_t)RB*Hd * 2);
  float*    hcol0 = (float*)   carve((size_t)RB*Hd * 4);
  float*    hcol1 = (float*)   carve((size_t)RB*Hd * 4);
  _Float16* hcol0h= (_Float16*)carve((size_t)RB*Hd * 2);
  _Float16* hcol1h= (_Float16*)carve((size_t)RB*Hd * 2);
  float*    g     = (float*)   carve((size_t)RB*1536 * 4);

  // weights -> f16 mirrors
  k_cvt_f16<<<(1536*768 + 255)/256, 256, 0, stream>>>(W,   Wh,   1536*768);
  k_cvt_f16<<<(384*128  + 255)/256, 256, 0, stream>>>(ipw, ipwh, 384*128);
  k_cvt_f16<<<(128*128  + 255)/256, 256, 0, stream>>>(opw, opwh, 128*128);

  // zero initial state (0x00 == 0.0f / 0.0h)
  hipMemsetAsync(hrow,   0, (size_t)RB*Hd*4, stream);
  hipMemsetAsync(hrowh,  0, (size_t)RB*Hd*2, stream);
  hipMemsetAsync(hcol0,  0, (size_t)RB*Hd*4, stream);
  hipMemsetAsync(hcol0h, 0, (size_t)RB*Hd*2, stream);

  // 1) build padded slice inputs (f16)
  k_build_x<<<(unsigned)(MALL*Id/256), 256, 0, stream>>>(inp, xh);

  // 2) qkv projection: (72192 x 128) @ (384 x 128)^T  -- 64x64 block tiles
  k_gemm_aw<<<(MALL/64) * (384/64), 128, 0, stream>>>(xh, ipwh, ipb, qkvh,
                                                      (int)MALL, 384, 128);
  // 3) attention scores + mask
  k_scores<<<Ld*Bn*2*4, 32, 0, stream>>>(qkvh, mask, sc);
  // 4) softmax
  k_softmax<<<Ld*Bn*2*24, 32, 0, stream>>>(sc, attnh);
  // 5) context = attn @ v
  k_ctx<<<Ld*Bn*2*8, 32, 0, stream>>>(attnh, qkvh, ch);
  // 6) output projection: (72192 x 128) @ (128 x 128)^T
  k_gemm_aw<<<(MALL/64) * (128/64), 128, 0, stream>>>(ch, opwh, opb, yh,
                                                      (int)MALL, 128, 128);

  // 7) sequential 2D-PSGMU recurrence: 47 steps of (1536x1536x768) WMMA GEMM + gates
  for (int l = 0; l < Ld; ++l) {
    const _Float16* xl = xh + (size_t)l * RB * Id;
    const _Float16* yl = yh + (size_t)l * RB * Id;
    float*    hc_in  = (l & 1) ? hcol1  : hcol0;
    float*    hc_out = (l & 1) ? hcol0  : hcol1;
    _Float16* hch_in = (l & 1) ? hcol1h : hcol0h;
    _Float16* hch_out= (l & 1) ? hcol0h : hcol1h;
    k_gemm_g<<<24*24, 128, 0, stream>>>(hrowh, hch_in, xl, yl, Wh, Bias, g, l);
    k_gates<<<RB*Hd/256, 256, 0, stream>>>(g, hrow, hc_in, hc_out, hrowh, hch_out, out, l);
  }
}
